// HMKGR_21861383536924
// MI455X (gfx1250) — matrix operand order repaired and checked by reading, not compiled, not run on hardware
//
#include <hip/hip_runtime.h>

#define N_ITEMS_C   50000
#define N_ENT_C     100000
#define N_USERS_C   50000
#define N_NODES_C   150000
#define EMBED       64
#define E_CKG_C     1200000
#define E_UKG_C     800000
#define IMG_DIM_C   1024
#define TXT_DIM_C   768
#define BATCH_C     4096
#define HIDDEN      256
#define SLOPE_C     0.01f
#define HSTRIDE     264   /* padded bf16 row stride (528B, 16B-aligned) for hidden tile */

typedef __attribute__((ext_vector_type(16))) __bf16        v16bf;
typedef __attribute__((ext_vector_type(8)))  float         v8f;
typedef __attribute__((ext_vector_type(4)))  unsigned int  u32x4;

struct BF16x16 { u32x4 lo, hi; };   // 32B carrier for one v16bf fragment

__device__ __forceinline__ __bf16 f2bf(float f) {
  unsigned u = __builtin_bit_cast(unsigned, f);
  u += 0x7FFFu + ((u >> 16) & 1u);            // round-to-nearest-even
  unsigned short h = (unsigned short)(u >> 16);
  return __builtin_bit_cast(__bf16, h);
}

// ---------------------------------------------------------------------------
// Pack a row-major f32 weight matrix W[K,N] into bf16 WMMA-B fragment order:
//   Wp[((kc*2 + h)*N + n)*16 + e]  holds  W[kc*32 + h*16 + e][n]
// so each lane's 16-element B fragment is 32 contiguous bytes.
// ---------------------------------------------------------------------------
__global__ void pack_w_kernel(const float* __restrict__ W, __bf16* __restrict__ Wp,
                              int K, int N) {
  long long i = (long long)blockIdx.x * blockDim.x + threadIdx.x;
  const long long stride = (long long)gridDim.x * blockDim.x;
  const long long total = (long long)K * N;
  for (; i < total; i += stride) {
    const int k = (int)(i / N), n = (int)(i % N);
    const int kc = k >> 5, rem = k & 31, h = rem >> 4, e = rem & 15;
    Wp[((((long long)kc * 2 + h) * N) + n) * 16 + e] = f2bf(W[i]);
  }
}

// ---------------------------------------------------------------------------
// Fused 2-layer MLP: out = leaky(X[50000,K1] @ W1[K1,256] + b1) @ W2[256,64] + b2
// 16 waves / block, 16-row tile per block, bf16 WMMA, f32 accumulate.
// Double-buffered bf16 A-staging in LDS; B fragments pre-packed in global.
// ---------------------------------------------------------------------------
__global__ __launch_bounds__(512)
void mlp2_wmma_kernel(const float* __restrict__ X, const __bf16* __restrict__ W1p,
                      const float* __restrict__ B1, const __bf16* __restrict__ W2p,
                      const float* __restrict__ B2, float* __restrict__ out, int K1) {
  __shared__ __align__(16) __bf16 a_stage[2][16 * 32];
  __shared__ __align__(16) __bf16 hid[16 * HSTRIDE];

  const int tid   = threadIdx.x;
  const int wv    = tid >> 5;
  const int lane  = tid & 31;
  const int row0  = blockIdx.x * 16;
  const int nm    = lane & 15;                 // M for A-frag, N for B/D-frag
  const int kbA   = (lane & 16) ? 16 : 0;      // A-fragment K base per half-wave
  const int hb    = (lane & 16) ? 1 : 0;       // B-fragment half selector
  const int mbase = (lane & 16) ? 8 : 0;       // D-fragment M base per half-wave
  const int sr = tid >> 5, sc = tid & 31;      // staging coords (512 thr = 16x32 tile)
  const int nk = K1 >> 5;

  // stage chunk 0 (f32 -> bf16, one element per thread)
  a_stage[0][sr * 32 + sc] = f2bf(X[(size_t)(row0 + sr) * K1 + sc]);
  __syncthreads();

  const int colH = wv * 16 + nm;               // hidden column owned by this lane
  v8f acc = {};
  for (int kc = 0; kc < nk; ++kc) {
    // prefetch next A tile into the other buffer
    if (kc + 1 < nk)
      a_stage[(kc + 1) & 1][sr * 32 + sc] =
          f2bf(X[(size_t)(row0 + sr) * K1 + ((kc + 1) << 5) + sc]);
    // A fragment: two contiguous 16B runs from LDS
    const __bf16* ap = &a_stage[kc & 1][nm * 32 + kbA];
    BF16x16 ta;
    ta.lo = *(const u32x4*)ap;
    ta.hi = *(const u32x4*)(ap + 8);
    const v16bf a = __builtin_bit_cast(v16bf, ta);
    // B fragment: 32 contiguous bytes from packed weights
    const __bf16* bp = W1p + (((size_t)(kc * 2 + hb) * HIDDEN) + colH) * 16;
    BF16x16 tb;
    tb.lo = *(const u32x4*)bp;
    tb.hi = *(const u32x4*)(bp + 8);
    const v16bf b = __builtin_bit_cast(v16bf, tb);
    acc = __builtin_amdgcn_wmma_f32_16x16x32_bf16(false, a, false, b,
                                                  (short)0, acc, false, false);
    __syncthreads();
  }

  // bias + leaky_relu -> bf16 hidden tile in LDS
  {
    const float bias = B1[colH];
#pragma unroll
    for (int r = 0; r < 8; ++r) {
      float v = acc[r] + bias;
      v = (v > 0.f) ? v : SLOPE_C * v;
      hid[(mbase + r) * HSTRIDE + colH] = f2bf(v);
    }
  }
  __syncthreads();

  // Layer 2: waves 0..3 cover N=64, K=256
  if (wv < 4) {
    const int colO = wv * 16 + nm;
    v8f acc2 = {};
#pragma unroll
    for (int kc = 0; kc < HIDDEN / 32; ++kc) {
      const __bf16* ap = &hid[nm * HSTRIDE + (kc << 5) + kbA];
      BF16x16 ta;
      ta.lo = *(const u32x4*)ap;
      ta.hi = *(const u32x4*)(ap + 8);
      const v16bf a = __builtin_bit_cast(v16bf, ta);
      const __bf16* bp = W2p + (((size_t)(kc * 2 + hb) * EMBED) + colO) * 16;
      BF16x16 tb;
      tb.lo = *(const u32x4*)bp;
      tb.hi = *(const u32x4*)(bp + 8);
      const v16bf b = __builtin_bit_cast(v16bf, tb);
      acc2 = __builtin_amdgcn_wmma_f32_16x16x32_bf16(false, a, false, b,
                                                     (short)0, acc2, false, false);
    }
    const float bias = B2[colO];
#pragma unroll
    for (int r = 0; r < 8; ++r)
      out[(size_t)(row0 + mbase + r) * EMBED + colO] = acc2[r] + bias;
  }
}

// ---------------------------------------------------------------------------
// Graph helpers (HBM / atomic bound)
// ---------------------------------------------------------------------------
__global__ void zero_kernel(float* __restrict__ p, long long n) {
  long long i = (long long)blockIdx.x * blockDim.x + threadIdx.x;
  const long long stride = (long long)gridDim.x * blockDim.x;
  for (; i < n; i += stride) p[i] = 0.f;
}

__global__ void copy_kernel(const float* __restrict__ s, float* __restrict__ d, long long n) {
  long long i = (long long)blockIdx.x * blockDim.x + threadIdx.x;
  const long long stride = (long long)gridDim.x * blockDim.x;
  for (; i < n; i += stride) d[i] = s[i];
}

__global__ void count_kernel(const int* __restrict__ heads, float* __restrict__ cnt, int E) {
  int i = blockIdx.x * blockDim.x + threadIdx.x;
  const int stride = gridDim.x * blockDim.x;
  for (; i < E; i += stride) atomicAdd(&cnt[heads[i]], 1.0f);
}

// msg = x[tail]*rel[type] scattered onto head; one lane per (edge, dim)
__global__ void scatter_kernel(const float* __restrict__ x, const int* __restrict__ ei,
                               const int* __restrict__ et, const float* __restrict__ rel,
                               float* __restrict__ outb, int E) {
  const long long total = (long long)E * EMBED;
  long long i = (long long)blockIdx.x * blockDim.x + threadIdx.x;
  const long long stride = (long long)gridDim.x * blockDim.x;
  for (; i < total; i += stride) {
    const long long e = i >> 6;
    const int d = (int)(i & 63);
    const int h = ei[e];
    const int t = ei[(long long)E + e];
    const int r = et[e];
    atomicAdd(&outb[(long long)h * EMBED + d],
              x[(long long)t * EMBED + d] * rel[r * EMBED + d]);
  }
}

// hop /= max(cnt,1); res += hop  (hop keeps normalized value for next hop)
__global__ void combine_kernel(float* __restrict__ hop, const float* __restrict__ cnt,
                               float* __restrict__ res, long long nnodes) {
  const long long total = nnodes * EMBED;
  long long i = (long long)blockIdx.x * blockDim.x + threadIdx.x;
  const long long stride = (long long)gridDim.x * blockDim.x;
  for (; i < total; i += stride) {
    const float c = fmaxf(cnt[i >> 6], 1.0f);
    const float v = hop[i] / c;
    hop[i] = v;
    res[i] += v;
  }
}

// ---------------------------------------------------------------------------
// Final gated fusion + dot + sigmoid, one block per (user,item) pair
// ---------------------------------------------------------------------------
__global__ __launch_bounds__(64)
void gate_dot_kernel(const float* __restrict__ fuI, const float* __restrict__ fuT,
                     const float* __restrict__ resI, const float* __restrict__ resT,
                     const float* __restrict__ g1w, const float* __restrict__ g1b,
                     const float* __restrict__ g2w, const float* __restrict__ g2b,
                     const float* __restrict__ g3w, const float* __restrict__ g3b,
                     const float* __restrict__ g4w, const float* __restrict__ g4b,
                     const int* __restrict__ user_ids, const int* __restrict__ item_ids,
                     float* __restrict__ out) {
  __shared__ float uig[EMBED], uia[EMBED], utg[EMBED], uta[EMBED], red[EMBED];
  const int b = blockIdx.x, t = threadIdx.x;
  const int u  = user_ids[b];
  const int it = item_ids[b];
  uig[t] = fuI [(size_t)u * EMBED + t];
  uia[t] = resI[((size_t)u + N_ENT_C) * EMBED + t];
  utg[t] = fuT [(size_t)u * EMBED + t];
  uta[t] = resT[((size_t)u + N_ENT_C) * EMBED + t];
  __syncthreads();
  float s1 = g1b[t] + g2b[t];
  float s2 = g3b[t] + g4b[t];
  for (int k = 0; k < EMBED; ++k) {
    s1 += uig[k] * g1w[k * EMBED + t] + uia[k] * g2w[k * EMBED + t];
    s2 += utg[k] * g3w[k * EMBED + t] + uta[k] * g4w[k * EMBED + t];
  }
  const float gi1 = 1.f / (1.f + expf(-s1));
  const float gi2 = 1.f / (1.f + expf(-s2));
  const float ue0 = gi1 * uig[t] + (1.f - gi1) * uia[t];
  const float ue1 = gi2 * utg[t] + (1.f - gi2) * uta[t];
  const float ie0 = resI[(size_t)it * EMBED + t];
  const float ie1 = resT[(size_t)it * EMBED + t];
  red[t] = ue0 * ie0 + ue1 * ie1;
  __syncthreads();
  for (int s = 32; s > 0; s >>= 1) {
    if (t < s) red[t] += red[t + s];
    __syncthreads();
  }
  if (t == 0) out[b] = 1.f / (1.f + expf(-red[0]));
}

// ---------------------------------------------------------------------------
extern "C" void kernel_launch(void* const* d_in, const int* in_sizes, int n_in,
                              void* d_out, int out_size, void* d_ws, size_t ws_size,
                              hipStream_t stream) {
  const float* image_features = (const float*)d_in[0];
  const float* text_features  = (const float*)d_in[1];
  const float* other_image    = (const float*)d_in[2];
  const float* other_text     = (const float*)d_in[3];
  const float* rel_image      = (const float*)d_in[4];
  const float* rel_text       = (const float*)d_in[5];
  const float* ukg_rel_image  = (const float*)d_in[6];
  const float* ukg_rel_text   = (const float*)d_in[7];
  const float* img_w1 = (const float*)d_in[8];
  const float* img_b1 = (const float*)d_in[9];
  const float* img_w2 = (const float*)d_in[10];
  const float* img_b2 = (const float*)d_in[11];
  const float* txt_w1 = (const float*)d_in[12];
  const float* txt_b1 = (const float*)d_in[13];
  const float* txt_w2 = (const float*)d_in[14];
  const float* txt_b2 = (const float*)d_in[15];
  const float* g1w = (const float*)d_in[16]; const float* g1b = (const float*)d_in[17];
  const float* g2w = (const float*)d_in[18]; const float* g2b = (const float*)d_in[19];
  const float* g3w = (const float*)d_in[20]; const float* g3b = (const float*)d_in[21];
  const float* g4w = (const float*)d_in[22]; const float* g4b = (const float*)d_in[23];
  const int* edge_index     = (const int*)d_in[24];
  const int* edge_type      = (const int*)d_in[25];
  const int* ukg_edge_index = (const int*)d_in[26];
  const int* ukg_edge_type  = (const int*)d_in[27];
  const int* user_ids       = (const int*)d_in[28];
  const int* item_ids       = (const int*)d_in[29];
  float* out = (float*)d_out;

  float* ws = (float*)d_ws;
  const long long NV = (long long)N_NODES_C * EMBED;   // 9.6M floats
  const long long UV = (long long)N_USERS_C * EMBED;   // 3.2M floats
  float* ego  = ws;               // ego embeddings / hop-2 scratch
  float* resI = ego  + NV;        // residual-accumulated image embeddings
  float* resT = resI + NV;        // residual-accumulated text embeddings
  float* tmpA = resT + NV;        // hop-1 scratch
  float* fuI  = tmpA + NV;        // user-KG image result
  float* fuT  = fuI  + UV;        // user-KG text result
  float* cnt  = fuT  + UV;        // CKG head degrees [N_NODES]
  float* cntU = cnt  + N_NODES_C; // UKG head degrees [N_USERS]

  // packed bf16 weights after the float region (32B-aligned)
  long long fend = 4 * NV + 2 * UV + N_NODES_C + N_USERS_C;
  fend = (fend + 7) & ~7LL;
  __bf16* w1pI = (__bf16*)(ws + fend);
  __bf16* w2pI = w1pI + (long long)IMG_DIM_C * HIDDEN;
  __bf16* w1pT = w2pI + (long long)HIDDEN * EMBED;
  __bf16* w2pT = w1pT + (long long)TXT_DIM_C * HIDDEN;

  const dim3 blk(256);
  auto blocks = [](long long n) -> dim3 {
    long long b = (n + 255) / 256;
    if (b > 262144) b = 262144;
    return dim3((unsigned)b);
  };

  // weight packing (bf16, WMMA B-fragment order)
  pack_w_kernel<<<blocks((long long)IMG_DIM_C * HIDDEN), blk, 0, stream>>>(img_w1, w1pI, IMG_DIM_C, HIDDEN);
  pack_w_kernel<<<blocks((long long)HIDDEN * EMBED),     blk, 0, stream>>>(img_w2, w2pI, HIDDEN, EMBED);
  pack_w_kernel<<<blocks((long long)TXT_DIM_C * HIDDEN), blk, 0, stream>>>(txt_w1, w1pT, TXT_DIM_C, HIDDEN);
  pack_w_kernel<<<blocks((long long)HIDDEN * EMBED),     blk, 0, stream>>>(txt_w2, w2pT, HIDDEN, EMBED);

  // degree counts (once)
  zero_kernel <<<blocks(N_NODES_C), blk, 0, stream>>>(cnt, N_NODES_C);
  count_kernel<<<blocks(E_CKG_C),   blk, 0, stream>>>(edge_index, cnt, E_CKG_C);
  zero_kernel <<<blocks(N_USERS_C), blk, 0, stream>>>(cntU, N_USERS_C);
  count_kernel<<<blocks(E_UKG_C),   blk, 0, stream>>>(ukg_edge_index, cntU, E_UKG_C);

  const long long OTH = (long long)(N_NODES_C - N_ITEMS_C) * EMBED;
  const long long EV  = (long long)E_CKG_C * EMBED;
  const long long UEV = (long long)E_UKG_C * EMBED;

  // ================= IMAGE modality over CKG =================
  mlp2_wmma_kernel<<<N_ITEMS_C / 16, 512, 0, stream>>>(image_features, w1pI, img_b1,
                                                       w2pI, img_b2, ego, IMG_DIM_C);
  copy_kernel<<<blocks(OTH), blk, 0, stream>>>(other_image, ego + (long long)N_ITEMS_C * EMBED, OTH);
  copy_kernel<<<blocks(NV), blk, 0, stream>>>(ego, resI, NV);
  zero_kernel<<<blocks(NV), blk, 0, stream>>>(tmpA, NV);
  scatter_kernel<<<blocks(EV), blk, 0, stream>>>(ego, edge_index, edge_type, rel_image, tmpA, E_CKG_C);
  combine_kernel<<<blocks(NV), blk, 0, stream>>>(tmpA, cnt, resI, N_NODES_C);
  zero_kernel<<<blocks(NV), blk, 0, stream>>>(ego, NV);
  scatter_kernel<<<blocks(EV), blk, 0, stream>>>(tmpA, edge_index, edge_type, rel_image, ego, E_CKG_C);
  combine_kernel<<<blocks(NV), blk, 0, stream>>>(ego, cnt, resI, N_NODES_C);

  // ================= TEXT modality over CKG =================
  mlp2_wmma_kernel<<<N_ITEMS_C / 16, 512, 0, stream>>>(text_features, w1pT, txt_b1,
                                                       w2pT, txt_b2, ego, TXT_DIM_C);
  copy_kernel<<<blocks(OTH), blk, 0, stream>>>(other_text, ego + (long long)N_ITEMS_C * EMBED, OTH);
  copy_kernel<<<blocks(NV), blk, 0, stream>>>(ego, resT, NV);
  zero_kernel<<<blocks(NV), blk, 0, stream>>>(tmpA, NV);
  scatter_kernel<<<blocks(EV), blk, 0, stream>>>(ego, edge_index, edge_type, rel_text, tmpA, E_CKG_C);
  combine_kernel<<<blocks(NV), blk, 0, stream>>>(tmpA, cnt, resT, N_NODES_C);
  zero_kernel<<<blocks(NV), blk, 0, stream>>>(ego, NV);
  scatter_kernel<<<blocks(EV), blk, 0, stream>>>(tmpA, edge_index, edge_type, rel_text, ego, E_CKG_C);
  combine_kernel<<<blocks(NV), blk, 0, stream>>>(ego, cnt, resT, N_NODES_C);

  // ================= USER KG, image =================
  const float* xuI = resI + (long long)N_ENT_C * EMBED;
  copy_kernel<<<blocks(UV), blk, 0, stream>>>(xuI, fuI, UV);
  zero_kernel<<<blocks(UV), blk, 0, stream>>>(tmpA, UV);
  scatter_kernel<<<blocks(UEV), blk, 0, stream>>>(xuI, ukg_edge_index, ukg_edge_type, ukg_rel_image, tmpA, E_UKG_C);
  combine_kernel<<<blocks(UV), blk, 0, stream>>>(tmpA, cntU, fuI, N_USERS_C);
  zero_kernel<<<blocks(UV), blk, 0, stream>>>(ego, UV);
  scatter_kernel<<<blocks(UEV), blk, 0, stream>>>(tmpA, ukg_edge_index, ukg_edge_type, ukg_rel_image, ego, E_UKG_C);
  combine_kernel<<<blocks(UV), blk, 0, stream>>>(ego, cntU, fuI, N_USERS_C);

  // ================= USER KG, text =================
  const float* xuT = resT + (long long)N_ENT_C * EMBED;
  copy_kernel<<<blocks(UV), blk, 0, stream>>>(xuT, fuT, UV);
  zero_kernel<<<blocks(UV), blk, 0, stream>>>(tmpA, UV);
  scatter_kernel<<<blocks(UEV), blk, 0, stream>>>(xuT, ukg_edge_index, ukg_edge_type, ukg_rel_text, tmpA, E_UKG_C);
  combine_kernel<<<blocks(UV), blk, 0, stream>>>(tmpA, cntU, fuT, N_USERS_C);
  zero_kernel<<<blocks(UV), blk, 0, stream>>>(ego, UV);
  scatter_kernel<<<blocks(UEV), blk, 0, stream>>>(tmpA, ukg_edge_index, ukg_edge_type, ukg_rel_text, ego, E_UKG_C);
  combine_kernel<<<blocks(UV), blk, 0, stream>>>(ego, cntU, fuT, N_USERS_C);

  // ================= gated fusion + score =================
  gate_dot_kernel<<<BATCH_C, EMBED, 0, stream>>>(fuI, fuT, resI, resT,
                                                 g1w, g1b, g2w, g2b, g3w, g3b, g4w, g4b,
                                                 user_ids, item_ids, out);
}
